// MoCo_66829691126519
// MI455X (gfx1250) — matrix operand order repaired and checked by reading, not compiled, not run on hardware
//
#include <hip/hip_runtime.h>
#include <hip/hip_bf16.h>

// ---------------------------------------------------------------------------
// Types for CDNA5 WMMA (wave32): V_WMMA_F32_16X16X32_BF16
// ---------------------------------------------------------------------------
typedef __attribute__((ext_vector_type(16))) __bf16 v16bf;
typedef __attribute__((ext_vector_type(8)))  float  v8f;

#define TEMP_INV (1.0f / 0.07f)

// Problem sizes (fixed by reference setup_inputs)
#define NB 4096   // batch
#define ND 256    // feature dim
#define NK 512    // codebook size
#define NL 32768  // queue_n rows
#define NQ 8192   // queue_k rows

// ---------------------------------------------------------------------------
// Counter-based RNG (JAX threefry is unverifiable here; hash gives iid draws)
// ---------------------------------------------------------------------------
__device__ __forceinline__ unsigned hash_u32(unsigned x) {
  x ^= x >> 16; x *= 0x7feb352dU; x ^= x >> 15; x *= 0x846ca68bU; x ^= x >> 16;
  return x;
}
__device__ __forceinline__ float u01(unsigned seed, unsigned idx) {
  unsigned h = hash_u32(seed ^ (idx * 0x9E3779B9u) ^ 0x85ebca6bu);
  return (float)(h >> 8) * (1.0f / 16777216.0f) + 1e-20f;
}
__device__ __forceinline__ float gumbelf(unsigned seed, unsigned idx) {
  return -__logf(-__logf(u01(seed, idx)));
}

// ---------------------------------------------------------------------------
// Small utility kernels
// ---------------------------------------------------------------------------
__global__ void k_zero(float* p, int n) {
  int i = blockIdx.x * blockDim.x + threadIdx.x;
  if (i < n) p[i] = 0.f;
}

__global__ __launch_bounds__(256) void k_sumsq(const float* __restrict__ x, long n,
                                               float* __restrict__ acc) {
  __shared__ float red[256];
  float s = 0.f;
  for (long i = blockIdx.x * (long)blockDim.x + threadIdx.x; i < n;
       i += (long)gridDim.x * blockDim.x) {
    float v = x[i];
    s += v * v;
  }
  red[threadIdx.x] = s; __syncthreads();
  for (int o = 128; o; o >>= 1) {
    if ((int)threadIdx.x < o) red[threadIdx.x] += red[threadIdx.x + o];
    __syncthreads();
  }
  if (!threadIdx.x) atomicAdd(acc, red[0]);
}

// dst = src * rsqrt(max(sumsq,1e-12)); optional f32 and/or bf16 destinations
__global__ __launch_bounds__(256) void k_l2n(const float* __restrict__ src,
                                             const float* __restrict__ ss,
                                             float* __restrict__ dstf,
                                             __bf16* __restrict__ dstb, long n) {
  float sc = rsqrtf(fmaxf(ss[0], 1e-12f));
  for (long i = blockIdx.x * (long)blockDim.x + threadIdx.x; i < n;
       i += (long)gridDim.x * blockDim.x) {
    float v = src[i] * sc;
    if (dstf) dstf[i] = v;
    if (dstb) dstb[i] = (__bf16)v;
  }
}

// queue = 2 * l2n(queue_k) + 0.1 * N(0,1); also accumulate sumsq(queue)
__global__ __launch_bounds__(256) void k_make_queue(const float* __restrict__ qk,
                                                    const float* __restrict__ ss_qk,
                                                    float* __restrict__ t,
                                                    float* __restrict__ ss_t, long n,
                                                    unsigned seed) {
  __shared__ float red[256];
  float sc = 2.f * rsqrtf(fmaxf(ss_qk[0], 1e-12f));
  float s = 0.f;
  for (long i = blockIdx.x * (long)blockDim.x + threadIdx.x; i < n;
       i += (long)gridDim.x * blockDim.x) {
    float u1 = u01(seed, (unsigned)(2 * i));
    float u2 = u01(seed, (unsigned)(2 * i + 1));
    float nr = sqrtf(-2.f * __logf(u1)) * __cosf(6.28318530718f * u2);
    float v = qk[i] * sc + 0.1f * nr;
    t[i] = v;
    s += v * v;
  }
  red[threadIdx.x] = s; __syncthreads();
  for (int o = 128; o; o >>= 1) {
    if ((int)threadIdx.x < o) red[threadIdx.x] += red[threadIdx.x + o];
    __syncthreads();
  }
  if (!threadIdx.x) atomicAdd(ss_t, red[0]);
}

// ---------------------------------------------------------------------------
// Tiled WMMA GEMM.  C[M,N] (f32) = A[M,Kd] * B, with
//   TRANSB=1: B supplied as Bsrc[N,Kd] row-major (C = A * Bsrc^T)
//   TRANSB=0: B supplied as Bsrc[Kd,N] row-major (C = A * Bsrc)
//   ABF=1   : A supplied as bf16 (Ab), else f32 (Af) converted on stage.
// Block = 256 threads = 8 wave32; block tile 128x64; wave tile 32x32 (2x2 WMMA).
// M%128==0, N%64==0, Kd%32==0 hold for all call sites.
// ---------------------------------------------------------------------------
template <bool TRANSB, bool ABF>
__global__ __launch_bounds__(256) void k_gemm(const float* __restrict__ Af,
                                              const __bf16* __restrict__ Ab,
                                              const float* __restrict__ Bsrc,
                                              float* __restrict__ C, int M, int N,
                                              int Kd) {
  // LDS: A tile 128x32 bf16 padded to lda=40 (bank-conflict-free: m*20 mod 64
  // distinct for m=0..15); B tile: NT [64][40], NN [32][72].
  __shared__ __bf16 As[128 * 40];
  __shared__ __bf16 Bs[2624];

  const int tid  = threadIdx.x;
  const int lane = tid & 31;
  const int w    = tid >> 5;
  const int wm   = (w >> 1) * 32;   // wave tile row in block tile (4 x 2 waves)
  const int wn   = (w & 1) * 32;
  const int m0   = blockIdx.y * 128;
  const int n0   = blockIdx.x * 64;
  const int mrow = lane & 15;
  const int hi   = lane >> 4;

  v8f acc[2][2] = {};

  for (int k0 = 0; k0 < Kd; k0 += 32) {
    __syncthreads();
    // ---- stage A (128x32) ----
    for (int i = tid; i < 128 * 32; i += 256) {
      int r = i >> 5, c = i & 31;
      size_t g = (size_t)(m0 + r) * Kd + k0 + c;
      As[r * 40 + c] = ABF ? Ab[g] : (__bf16)Af[g];
    }
    // ---- stage B ----
    if (TRANSB) {
      for (int i = tid; i < 64 * 32; i += 256) {
        int r = i >> 5, c = i & 31;
        Bs[r * 40 + c] = (__bf16)Bsrc[(size_t)(n0 + r) * Kd + k0 + c];
      }
    } else {
      for (int i = tid; i < 32 * 64; i += 256) {
        int r = i >> 6, c = i & 63;
        Bs[r * 72 + c] = (__bf16)Bsrc[(size_t)(k0 + r) * N + n0 + c];
      }
    }
    // prefetch next K-tile of A into L2 (global_prefetch_b8)
    if (k0 + 32 < Kd) {
      size_t g2 = (size_t)(m0 + (tid >> 1)) * Kd + k0 + 32 + (tid & 1) * 16;
      if (ABF) __builtin_prefetch(Ab + g2, 0, 0);
      else     __builtin_prefetch(Af + g2, 0, 0);
    }
    __syncthreads();

    // ---- fragments: contiguous k = 16*(l>=16) + e, same order in A and B ----
    v16bf af[2], bf[2];
#pragma unroll
    for (int mi = 0; mi < 2; ++mi) {
      // A 16x32 bf16: lane m=l&15; K = base + e,
      // base = 16*(l>=16)
      const __bf16* ap = &As[(wm + mi * 16 + mrow) * 40 + hi * 16];
#pragma unroll
      for (int e = 0; e < 16; ++e) af[mi][e] = ap[e];
    }
#pragma unroll
    for (int ni = 0; ni < 2; ++ni) {
      // B 32x16: lane holds column n=l&15; K = 16*(l>=16) + e
      if (TRANSB) {
        const __bf16* bp = &Bs[(wn + ni * 16 + mrow) * 40 + hi * 16];
#pragma unroll
        for (int e = 0; e < 16; ++e) bf[ni][e] = bp[e];
      } else {
        const __bf16* bp = &Bs[(hi * 16) * 72 + wn + ni * 16 + mrow];
#pragma unroll
        for (int e = 0; e < 16; ++e) bf[ni][e] = bp[e * 72];
      }
    }
#pragma unroll
    for (int mi = 0; mi < 2; ++mi)
#pragma unroll
      for (int ni = 0; ni < 2; ++ni)
        acc[mi][ni] = __builtin_amdgcn_wmma_f32_16x16x32_bf16(
            false, af[mi], false, bf[ni], (short)0, acc[mi][ni], false, false);
  }

  // ---- store C (f32 C/D layout: row = v + 8*(l>=16), col = l&15) ----
#pragma unroll
  for (int mi = 0; mi < 2; ++mi)
#pragma unroll
    for (int ni = 0; ni < 2; ++ni)
#pragma unroll
      for (int v = 0; v < 8; ++v) {
        int row = m0 + wm + mi * 16 + v + 8 * hi;
        int col = n0 + wn + ni * 16 + mrow;
        C[(size_t)row * N + col] = acc[mi][ni][v];
      }
}

// ---------------------------------------------------------------------------
// Gumbel softmax over one row of logits[B,K] (K=512), + hard argmax index.
// ---------------------------------------------------------------------------
__global__ __launch_bounds__(256) void k_gumbel_softmax(
    const float* __restrict__ logits, __bf16* __restrict__ assign,
    float* __restrict__ out_idx, unsigned seed_soft, unsigned seed_hard) {
  const int K = NK;
  int b = blockIdx.x, t = threadIdx.x;
  __shared__ float red[256];
  __shared__ int redi[256];

  float l0 = logits[(size_t)b * K + t];
  float l1 = logits[(size_t)b * K + t + 256];
  float t0 = l0 + gumbelf(seed_soft, (unsigned)(b * K + t));
  float t1 = l1 + gumbelf(seed_soft, (unsigned)(b * K + t + 256));

  red[t] = fmaxf(t0, t1); __syncthreads();
  for (int o = 128; o; o >>= 1) {
    if (t < o) red[t] = fmaxf(red[t], red[t + o]);
    __syncthreads();
  }
  float mx = red[0]; __syncthreads();
  float e0 = __expf(t0 - mx), e1 = __expf(t1 - mx);
  red[t] = e0 + e1; __syncthreads();
  for (int o = 128; o; o >>= 1) {
    if (t < o) red[t] += red[t + o];
    __syncthreads();
  }
  float inv = 1.f / red[0];
  assign[(size_t)b * K + t]       = (__bf16)(e0 * inv);
  assign[(size_t)b * K + t + 256] = (__bf16)(e1 * inv);

  if (out_idx) {  // hard assignment = argmax(logits + gumbel2)
    float h0 = l0 + gumbelf(seed_hard, (unsigned)(b * K + t));
    float h1 = l1 + gumbelf(seed_hard, (unsigned)(b * K + t + 256));
    float hv; int hidx;
    if (h0 >= h1) { hv = h0; hidx = t; } else { hv = h1; hidx = t + 256; }
    __syncthreads();
    red[t] = hv; redi[t] = hidx; __syncthreads();
    for (int o = 128; o; o >>= 1) {
      if (t < o && red[t + o] > red[t]) { red[t] = red[t + o]; redi[t] = redi[t + o]; }
      __syncthreads();
    }
    if (!t) out_idx[b] = (float)redi[0];
  }
}

// ---------------------------------------------------------------------------
// Stochastic binarization of kn=logits^T: bin[k,b] = sigmoid(kn) > U(0,1).
// One block per codebook row k; also produces row sums.
// ---------------------------------------------------------------------------
__global__ __launch_bounds__(256) void k_binary(const float* __restrict__ logits,
                                                __bf16* __restrict__ bin,
                                                float* __restrict__ rowsum,
                                                unsigned seed) {
  const int K = NK, Bn = NB;
  int k = blockIdx.x, t = threadIdx.x;
  __shared__ float red[256];
  float s = 0.f;
  for (int b = t; b < Bn; b += 256) {
    float kn = logits[(size_t)b * K + k];
    float p  = 1.f / (1.f + __expf(-kn));
    float v  = (p > u01(seed, (unsigned)(k * Bn + b))) ? 1.f : 0.f;
    bin[(size_t)k * Bn + b] = (__bf16)v;
    s += v;
  }
  red[t] = s; __syncthreads();
  for (int o = 128; o; o >>= 1) {
    if (t < o) red[t] += red[t + o];
    __syncthreads();
  }
  if (!t) rowsum[k] = red[0];
}

__global__ void k_rowdiv(float* __restrict__ agg, const float* __restrict__ rowsum) {
  int k = blockIdx.x;
  float inv = 1.f / (rowsum[k] + 1e-8f);
  agg[(size_t)k * ND + threadIdx.x] *= inv;  // blockDim = ND = 256
}

__global__ __launch_bounds__(256) void k_lpos(const float* __restrict__ q,
                                              const float* __restrict__ kp,
                                              float* __restrict__ lpos) {
  int r = blockIdx.x, t = threadIdx.x;
  __shared__ float red[256];
  red[t] = q[(size_t)r * ND + t] * kp[(size_t)r * ND + t];
  __syncthreads();
  for (int o = 128; o; o >>= 1) {
    if (t < o) red[t] += red[t + o];
    __syncthreads();
  }
  if (!t) lpos[r] = red[0] * TEMP_INV;
}

// ---------------------------------------------------------------------------
// Fused InfoNCE helpers
// ---------------------------------------------------------------------------
// Branchless online logsumexp update (v_cndmask, no EXEC churn):
// e = exp(-|x-m|);  s = (x>m) ? s*e+1 : s+e;  m = max(m,x)
__device__ __forceinline__ void lse_update(float x, float& m, float& s) {
  float d = x - m;
  float e = __expf(-fabsf(d));
  bool  g = d > 0.f;
  s = g ? __builtin_fmaf(s, e, 1.f) : (s + e);
  m = g ? x : m;
}

__device__ __forceinline__ v16bf load_bfrag(const __bf16* __restrict__ negs,
                                            int col, int s, int mrow, int hi) {
  // B 32x16 bf16 layout: lane holds column n = lane&15; K = 16*(lane>=16) + e
  const __bf16* bp = negs + (size_t)(col + mrow) * ND + s * 32 + hi * 16;
  v16bf b;
#pragma unroll
  for (int e = 0; e < 16; ++e) b[e] = bp[e];
  return b;
}

// ---------------------------------------------------------------------------
// Fused InfoNCE: per-row logsumexp over [lpos, q @ negs^T / temp] without ever
// materializing the [N, L] logit matrix.  Block handles 16 rows; 8 waves split
// the L range; 32 columns per iteration (two WMMA accumulator chains) with
// double-buffered B fragments so loads for k-step s+1 overlap the WMMAs of
// step s.  negs streams from L2 (16 MB << 192 MB, reused by 256 row-blocks).
// ---------------------------------------------------------------------------
__global__ __launch_bounds__(256) void k_nce(const __bf16* __restrict__ Qm,
                                             const __bf16* __restrict__ negs,
                                             const float* __restrict__ lpos, int L,
                                             float* __restrict__ loss_acc) {
  const int D = ND;
  const int m0   = blockIdx.x * 16;
  const int lane = threadIdx.x & 31;
  const int w    = threadIdx.x >> 5;
  const int mrow = lane & 15;
  const int hi   = lane >> 4;

  // Preload the 16x256 A strip as 8 K-fragments (stays in VGPRs all kernel).
  v16bf afr[8];
#pragma unroll
  for (int s = 0; s < 8; ++s) {
    const __bf16* ap = Qm + (size_t)(m0 + mrow) * D + s * 32 + hi * 16;
#pragma unroll
    for (int e = 0; e < 16; ++e) afr[s][e] = ap[e];
  }

  float rm[8], rs[8];
#pragma unroll
  for (int v = 0; v < 8; ++v) { rm[v] = -1e30f; rs[v] = 0.f; }

  const int per = L / 8;  // columns per wave (32768/8=4096, 8192/8=1024; %32==0)
  for (int c0 = w * per; c0 < (w + 1) * per; c0 += 32) {
    v8f acc0 = {}, acc1 = {};
    v16bf b0 = load_bfrag(negs, c0, 0, mrow, hi);
    v16bf b1 = load_bfrag(negs, c0 + 16, 0, mrow, hi);
    v16bf b0n, b1n;
#pragma unroll
    for (int s = 0; s < 8; ++s) {
      if (s < 7) {  // issue next k-step's loads before consuming current frags
        b0n = load_bfrag(negs, c0, s + 1, mrow, hi);
        b1n = load_bfrag(negs, c0 + 16, s + 1, mrow, hi);
      }
      acc0 = __builtin_amdgcn_wmma_f32_16x16x32_bf16(false, afr[s], false, b0,
                                                     (short)0, acc0, false, false);
      acc1 = __builtin_amdgcn_wmma_f32_16x16x32_bf16(false, afr[s], false, b1,
                                                     (short)0, acc1, false, false);
      if (s < 7) { b0 = b0n; b1 = b1n; }
    }
#pragma unroll
    for (int v = 0; v < 8; ++v) {
      lse_update(acc0[v] * TEMP_INV, rm[v], rs[v]);
      lse_update(acc1[v] * TEMP_INV, rm[v], rs[v]);
    }
  }

  // Reduce across the 16 lanes holding different columns of the same rows.
#pragma unroll
  for (int off = 1; off < 16; off <<= 1) {
#pragma unroll
    for (int v = 0; v < 8; ++v) {
      float om = __shfl_xor(rm[v], off, 32);
      float os = __shfl_xor(rs[v], off, 32);
      float M  = fmaxf(rm[v], om);
      rs[v] = rs[v] * __expf(rm[v] - M) + os * __expf(om - M);
      rm[v] = M;
    }
  }
  __shared__ float lm[8][16], ls[8][16];
  if (mrow == 0) {
#pragma unroll
    for (int v = 0; v < 8; ++v) { lm[w][hi * 8 + v] = rm[v]; ls[w][hi * 8 + v] = rs[v]; }
  }
  __syncthreads();
  if (threadIdx.x < 16) {
    int r = threadIdx.x;
    float lp = lpos[m0 + r];
    float m = lp, s = 1.f;  // seed with the positive logit
    for (int ww = 0; ww < 8; ++ww) {
      float om = lm[ww][r], os = ls[ww][r];
      float M  = fmaxf(m, om);
      s = s * __expf(m - M) + os * __expf(om - M);
      m = M;
    }
    atomicAdd(loss_acc, (m + __logf(s)) - lp);  // logsumexp - positive
  }
}

__global__ void k_final(const float* ln, const float* lk, float* out) {
  if (threadIdx.x == 0 && blockIdx.x == 0)
    out[0] = ln[0] / (float)NB + lk[0] / (float)NK;
}

// ---------------------------------------------------------------------------
// Host-side orchestration (graph-capture safe: only kernel launches on stream)
// ---------------------------------------------------------------------------
extern "C" void kernel_launch(void* const* d_in, const int* in_sizes, int n_in,
                              void* d_out, int out_size, void* d_ws, size_t ws_size,
                              hipStream_t stream) {
  const float* feat1 = (const float*)d_in[0];
  const float* feat2 = (const float*)d_in[1];
  const float* ctx1  = (const float*)d_in[2];
  const float* ctx2  = (const float*)d_in[3];
  const float* qn    = (const float*)d_in[4];
  const float* qk    = (const float*)d_in[5];

  float* out = (float*)d_out;
  float* out_assign = out;                       // [NB]
  float* out_aggn2  = out + NB;                  // [NB*ND]
  float* out_aggk2  = out + NB + NB * ND;        // [NK*ND]
  float* out_loss   = out + NB + NB * ND + NK * ND;

  // Workspace carve-up
  char* wsb = (char*)d_ws;
  size_t off = 0;
  auto alloc = [&](size_t bytes) -> void* {
    off = (off + 255) & ~(size_t)255;
    void* p = wsb + off;
    off += bytes;
    return p;
  };
  float*  logits1  = (float*)alloc((size_t)NB * NK * 4);
  float*  logits2  = (float*)alloc((size_t)NB * NK * 4);
  __bf16* assign1  = (__bf16*)alloc((size_t)NB * NK * 2);
  __bf16* assign2  = (__bf16*)alloc((size_t)NB * NK * 2);
  __bf16* bin1     = (__bf16*)alloc((size_t)NK * NB * 2);
  __bf16* bin2     = (__bf16*)alloc((size_t)NK * NB * 2);
  __bf16* qn_b     = (__bf16*)alloc((size_t)NL * ND * 2);
  __bf16* qq_b     = (__bf16*)alloc((size_t)NQ * ND * 2);
  float*  tq       = (float*)alloc((size_t)NQ * ND * 4);
  float*  aggn1    = (float*)alloc((size_t)NB * ND * 4);
  float*  aggn2raw = (float*)alloc((size_t)NB * ND * 4);
  __bf16* aggn1_b  = (__bf16*)alloc((size_t)NB * ND * 2);
  float*  aggk1    = (float*)alloc((size_t)NK * ND * 4);
  float*  aggk2raw = (float*)alloc((size_t)NK * ND * 4);
  __bf16* aggk1_b  = (__bf16*)alloc((size_t)NK * ND * 2);
  float*  lpos_n   = (float*)alloc((size_t)NB * 4);
  float*  lpos_k   = (float*)alloc((size_t)NK * 4);
  float*  rowsum1  = (float*)alloc((size_t)NK * 4);
  float*  rowsum2  = (float*)alloc((size_t)NK * 4);
  float*  scal     = (float*)alloc(16 * 4);
  // scal: 0=ss_qn 1=ss_qk 2=ss_queue 3=ss_n1 4=ss_n2 5=ss_k1 6=ss_k2 7=loss_n 8=loss_k

  k_zero<<<1, 64, 0, stream>>>(scal, 16);

  // --- queue normalization (global l2n) + noisy queue for loss_k ---
  k_sumsq<<<2048, 256, 0, stream>>>(qn, (long)NL * ND, scal + 0);
  k_sumsq<<<1024, 256, 0, stream>>>(qk, (long)NQ * ND, scal + 1);
  k_l2n<<<2048, 256, 0, stream>>>(qn, scal + 0, nullptr, qn_b, (long)NL * ND);
  k_make_queue<<<1024, 256, 0, stream>>>(qk, scal + 1, tq, scal + 2, (long)NQ * ND,
                                         0xA5A51u);
  k_l2n<<<1024, 256, 0, stream>>>(tq, scal + 2, nullptr, qq_b, (long)NQ * ND);

  // --- kn^T = feat @ ctx^T : [NB, NK] ---
  k_gemm<true, false><<<dim3(NK / 64, NB / 128), 256, 0, stream>>>(
      feat1, nullptr, ctx1, logits1, NB, NK, ND);
  k_gemm<true, false><<<dim3(NK / 64, NB / 128), 256, 0, stream>>>(
      feat2, nullptr, ctx2, logits2, NB, NK, ND);

  // --- gumbel softmax (soft + hard argmax) ---
  k_gumbel_softmax<<<NB, 256, 0, stream>>>(logits1, assign1, out_assign, 0x111u, 0x222u);
  k_gumbel_softmax<<<NB, 256, 0, stream>>>(logits2, assign2, nullptr, 0x333u, 0x444u);

  // --- stochastic binary K->B assignment ---
  k_binary<<<NK, 256, 0, stream>>>(logits1, bin1, rowsum1, 0x555u);
  k_binary<<<NK, 256, 0, stream>>>(logits2, bin2, rowsum2, 0x666u);

  // --- agg_n = assign @ ctx : [NB, ND] ; agg_k = bin @ feat : [NK, ND] ---
  k_gemm<false, true><<<dim3(ND / 64, NB / 128), 256, 0, stream>>>(
      nullptr, assign1, ctx1, aggn1, NB, ND, NK);
  k_gemm<false, true><<<dim3(ND / 64, NB / 128), 256, 0, stream>>>(
      nullptr, assign2, ctx2, aggn2raw, NB, ND, NK);
  k_gemm<false, true><<<dim3(ND / 64, NK / 128), 256, 0, stream>>>(
      nullptr, bin1, feat1, aggk1, NK, ND, NB);
  k_gemm<false, true><<<dim3(ND / 64, NK / 128), 256, 0, stream>>>(
      nullptr, bin2, feat2, aggk2raw, NK, ND, NB);
  k_rowdiv<<<NK, ND, 0, stream>>>(aggk1, rowsum1);
  k_rowdiv<<<NK, ND, 0, stream>>>(aggk2raw, rowsum2);

  // --- global l2n of the four aggregates ---
  k_sumsq<<<1024, 256, 0, stream>>>(aggn1, (long)NB * ND, scal + 3);
  k_sumsq<<<1024, 256, 0, stream>>>(aggn2raw, (long)NB * ND, scal + 4);
  k_sumsq<<<256, 256, 0, stream>>>(aggk1, (long)NK * ND, scal + 5);
  k_sumsq<<<256, 256, 0, stream>>>(aggk2raw, (long)NK * ND, scal + 6);
  k_l2n<<<1024, 256, 0, stream>>>(aggn1, scal + 3, aggn1, aggn1_b, (long)NB * ND);
  k_l2n<<<1024, 256, 0, stream>>>(aggn2raw, scal + 4, out_aggn2, nullptr, (long)NB * ND);
  k_l2n<<<256, 256, 0, stream>>>(aggk1, scal + 5, aggk1, aggk1_b, (long)NK * ND);
  k_l2n<<<256, 256, 0, stream>>>(aggk2raw, scal + 6, out_aggk2, nullptr, (long)NK * ND);

  // --- InfoNCE: positives, then fused GEMM+logsumexp over the negatives ---
  k_lpos<<<NB, 256, 0, stream>>>(aggn1, out_aggn2, lpos_n);
  k_lpos<<<NK, 256, 0, stream>>>(aggk1, out_aggk2, lpos_k);
  k_nce<<<NB / 16, 256, 0, stream>>>(aggn1_b, qn_b, lpos_n, NL, scal + 7);
  k_nce<<<NK / 16, 256, 0, stream>>>(aggk1_b, qq_b, lpos_k, NQ, scal + 8);

  k_final<<<1, 32, 0, stream>>>(scal + 7, scal + 8, out_loss);
}
